// PM_3831110828272
// MI455X (gfx1250) — matrix operand order repaired and checked by reading, not compiled, not run on hardware
//
#include <hip/hip_runtime.h>

// ---------------------------------------------------------------------------
// CDNA5 (gfx1250) wave32 WMMA implementation of the PM fusion module.
// All GEMM-shaped work (convs via implicit GEMM, QK^T energy, attention*V)
// goes through v_wmma_f32_16x16x32_bf16 with f32 accumulation.
//
// This revision:
//  - conv kernel templated on (KS,PAD,DIL) with H=W=32 hard-coded: all pixel
//    decodes are shifts/masks (no runtime integer division sequences).
//  - LDS tiles whose K axis feeds WMMA fragments are stored pair-interleaved
//    ([k>>1][2n+(k&1)]) so every fragment element-pair is one aligned
//    ds_load_b32 (u32x8 -> bf16x16 bit-cast).
// ---------------------------------------------------------------------------

typedef __attribute__((ext_vector_type(16))) __bf16    bf16x16;
typedef __attribute__((ext_vector_type(8)))  float     f32x8;
typedef __attribute__((ext_vector_type(8)))  unsigned  u32x8;

__device__ __forceinline__ unsigned short f2bf(float f) {
    unsigned u = __builtin_bit_cast(unsigned, f);
    u += 0x7FFFu + ((u >> 16) & 1u);   // round-to-nearest-even
    return (unsigned short)(u >> 16);
}
__device__ __forceinline__ float bf2f(unsigned short h) {
    unsigned u = ((unsigned)h) << 16;
    return __builtin_bit_cast(float, u);
}

// ---------------------------------------------------------------------------
// Implicit-GEMM conv: Y[b,co,h,w] = act( scale[co] * (sum W*X) + shift[co] )
// X  : bf16 (B,Cin,32,32) ; Wt : bf16 packed (KS*KS, Cout, Cin)
// Block tile: 64 Cout rows x 64 pixels; 8 waves (4 row x 2 col), 2 accs/wave.
// ---------------------------------------------------------------------------
template <int KS, int PAD, int DIL>
__global__ void __launch_bounds__(256) conv_wmma_kernel(
    const unsigned short* __restrict__ X,
    const unsigned short* __restrict__ Wt,
    const float* __restrict__ sc, const float* __restrict__ sh,
    float* __restrict__ Yf, unsigned short* __restrict__ Yb,
    int Cin, int Cout, int relu)
{
    constexpr int H = 32, W = 32, HW = 1024;
    const int n0  = blockIdx.x * 64;
    const int co0 = blockIdx.y * 64;
    const int t    = threadIdx.x;
    const int lane = t & 31, wv = t >> 5;
    const int wm = wv & 3, wn = wv >> 2;

    __shared__ alignas(16) unsigned short As[64][40];    // [cout_local][k]
    __shared__ alignas(16) unsigned short Bs[16][136];   // pair-interleaved [k>>1][2n+(k&1)]

    f32x8 acc0 = {0.f,0.f,0.f,0.f,0.f,0.f,0.f,0.f};
    f32x8 acc1 = acc0;

    const int ar = t >> 2;          // A row 0..63
    const int ac = (t & 3) << 3;    // A col 0/8/16/24
    const int bk = t >> 3;          // B k 0..31
    const int bn = (t & 7) << 3;    // B pixel 0..56
    const int bkp = bk >> 1, bko = bk & 1;

    int pb[8], ph[8], pw[8];
#pragma unroll
    for (int j = 0; j < 8; ++j) {
        int p = n0 + bn + j;
        pb[j] = p >> 10;
        ph[j] = (p >> 5) & 31;
        pw[j] = p & 31;
    }

    // fragment geometry (constant per thread)
    const int arow = wm * 16 + (lane & 15);
    const int aks  = (lane < 16) ? 0 : 16;
    const int bcol = (lane & 15) + wn * 32;
    const int bkb2 = (lane < 16) ? 0 : 8;      // (k-base 0/16) >> 1

#pragma unroll
    for (int kh = 0; kh < KS; ++kh) {
#pragma unroll
        for (int kw = 0; kw < KS; ++kw) {
            const int dh = kh * DIL - PAD, dw = kw * DIL - PAD;
            const int tap = kh * KS + kw;
            for (int ci0 = 0; ci0 < Cin; ci0 += 32) {
                __syncthreads();
                // ---- A tile: weights (128-bit loads; zero rows >= Cout)
                if (co0 + ar < Cout) {
                    const unsigned short* src =
                        Wt + ((long)tap * Cout + (co0 + ar)) * Cin + ci0 + ac;
                    *(uint4*)&As[ar][ac] = *(const uint4*)src;
                    if (ci0 + 32 < Cin)
                        __builtin_prefetch((const void*)(src + 32), 0, 1);
                } else {
                    uint4 z = {0u, 0u, 0u, 0u};
                    *(uint4*)&As[ar][ac] = z;
                }
                // ---- B tile: im2col gather, pair-interleaved store
#pragma unroll
                for (int j = 0; j < 8; ++j) {
                    int ih = ph[j] + dh, iw = pw[j] + dw;
                    unsigned short v = 0;
                    if ((unsigned)ih < (unsigned)H && (unsigned)iw < (unsigned)W)
                        v = X[(((long)pb[j] * Cin + (ci0 + bk)) << 10) + (ih << 5) + iw];
                    Bs[bkp][((bn + j) << 1) | bko] = v;
                }
                __syncthreads();

                // ---- fragments: one ds_load_b32 per K-pair
                u32x8 au, bu0, bu1;
#pragma unroll
                for (int v = 0; v < 8; ++v) {
                    int k0 = (v << 1) + aks;
                    au[v] = *(const unsigned*)&As[arow][k0];
                }
#pragma unroll
                for (int v = 0; v < 8; ++v) {
                    bu0[v] = *(const unsigned*)&Bs[bkb2 + v][bcol << 1];
                    bu1[v] = *(const unsigned*)&Bs[bkb2 + v][(bcol << 1) + 32];
                }
                bf16x16 a  = __builtin_bit_cast(bf16x16, au);
                bf16x16 b0 = __builtin_bit_cast(bf16x16, bu0);
                bf16x16 b1 = __builtin_bit_cast(bf16x16, bu1);
                acc0 = __builtin_amdgcn_wmma_f32_16x16x32_bf16(
                           false, a, false, b0, (short)0, acc0, false, false);
                acc1 = __builtin_amdgcn_wmma_f32_16x16x32_bf16(
                           false, a, false, b1, (short)0, acc1, false, false);
            }
        }
    }

    // ---- epilogue: fused bias/BN/ReLU, f32 and/or bf16 outputs
    const int col = lane & 15;
    const int mo  = (lane < 16) ? 0 : 8;
#pragma unroll
    for (int r = 0; r < 8; ++r) {
        int m = co0 + wm * 16 + mo + r;
        if (m >= Cout) continue;
        float s  = sc ? sc[m] : 1.f;
        float b_ = sh ? sh[m] : 0.f;
#pragma unroll
        for (int h2 = 0; h2 < 2; ++h2) {
            int p   = n0 + wn * 32 + h2 * 16 + col;
            float v = (h2 ? acc1[r] : acc0[r]) * s + b_;
            if (relu) v = fmaxf(v, 0.f);
            long o = (((long)(p >> 10) * Cout + m) << 10) + (p & 1023);
            if (Yf) Yf[o] = v;
            if (Yb) Yb[o] = f2bf(v);
        }
    }
}

// ---------------------------------------------------------------------------
// energy[b,i,j] = wr[b]*sum_c qr[b,c,i]*kr[b,c,j] + wd[b]*sum_c qd*kd ; K=32
// All four 32x64 tiles stored pair-interleaved: element (k,idx) at
// [k>>1][2*idx+(k&1)] -> both A-side (transposed) and B-side fragment pairs
// are single aligned b32 LDS reads.
// ---------------------------------------------------------------------------
__global__ void __launch_bounds__(256) qk_wmma_kernel(
    const unsigned short* __restrict__ qr, const unsigned short* __restrict__ kr,
    const unsigned short* __restrict__ qd, const unsigned short* __restrict__ kd,
    const float* __restrict__ wsm, float* __restrict__ E)
{
    constexpr int N = 1024;
    const int b  = blockIdx.z;
    const int i0 = blockIdx.y * 64, j0 = blockIdx.x * 64;
    const int t  = threadIdx.x;
    const int lane = t & 31, wv = t >> 5, wm = wv & 3, wn = wv >> 2;

    __shared__ alignas(16) unsigned short Qr[16][136], Kr[16][136];
    __shared__ alignas(16) unsigned short Qd[16][136], Kd[16][136];

    const int kk = t >> 3, mb = (t & 7) << 3;
    const int kp = kk >> 1, ko = kk & 1;
    {
        long oq = ((long)b * 32 + kk) * N;
        uint4 vq = *(const uint4*)(qr + oq + i0 + mb);
        uint4 vk = *(const uint4*)(kr + oq + j0 + mb);
        uint4 vs = *(const uint4*)(qd + oq + i0 + mb);
        uint4 vt = *(const uint4*)(kd + oq + j0 + mb);
        const unsigned short* pq = (const unsigned short*)&vq;
        const unsigned short* pk = (const unsigned short*)&vk;
        const unsigned short* ps = (const unsigned short*)&vs;
        const unsigned short* pt = (const unsigned short*)&vt;
#pragma unroll
        for (int j = 0; j < 8; ++j) {
            int c = ((mb + j) << 1) | ko;
            Qr[kp][c] = pq[j];
            Kr[kp][c] = pk[j];
            Qd[kp][c] = ps[j];
            Kd[kp][c] = pt[j];
        }
    }
    __syncthreads();

    const int row  = wm * 16 + (lane & 15);
    const int ks2  = (lane < 16) ? 0 : 8;      // (k-base 0/16) >> 1 for A side
    const int col  = (lane & 15) + wn * 32;
    const int kb2  = (lane < 16) ? 0 : 8;      // (k-base 0/16) >> 1 for B side

    u32x8 aRu, aDu, br0u, br1u, bd0u, bd1u;
#pragma unroll
    for (int v = 0; v < 8; ++v) {
        int kp2 = v + ks2;   // pair row of k0
        aRu[v] = *(const unsigned*)&Qr[kp2][row << 1];
        aDu[v] = *(const unsigned*)&Qd[kp2][row << 1];
        br0u[v] = *(const unsigned*)&Kr[kb2 + v][col << 1];
        br1u[v] = *(const unsigned*)&Kr[kb2 + v][(col << 1) + 32];
        bd0u[v] = *(const unsigned*)&Kd[kb2 + v][col << 1];
        bd1u[v] = *(const unsigned*)&Kd[kb2 + v][(col << 1) + 32];
    }
    bf16x16 aR  = __builtin_bit_cast(bf16x16, aRu);
    bf16x16 aD  = __builtin_bit_cast(bf16x16, aDu);
    bf16x16 br0 = __builtin_bit_cast(bf16x16, br0u);
    bf16x16 br1 = __builtin_bit_cast(bf16x16, br1u);
    bf16x16 bd0 = __builtin_bit_cast(bf16x16, bd0u);
    bf16x16 bd1 = __builtin_bit_cast(bf16x16, bd1u);

    f32x8 z = {0.f,0.f,0.f,0.f,0.f,0.f,0.f,0.f};
    f32x8 aR0 = __builtin_amdgcn_wmma_f32_16x16x32_bf16(false, aR, false, br0, (short)0, z, false, false);
    f32x8 aR1 = __builtin_amdgcn_wmma_f32_16x16x32_bf16(false, aR, false, br1, (short)0, z, false, false);
    f32x8 aD0 = __builtin_amdgcn_wmma_f32_16x16x32_bf16(false, aD, false, bd0, (short)0, z, false, false);
    f32x8 aD1 = __builtin_amdgcn_wmma_f32_16x16x32_bf16(false, aD, false, bd1, (short)0, z, false, false);

    const float wr = wsm[b], wd = wsm[8 + b];
    const int c2 = lane & 15;
    const int mo = (lane < 16) ? 0 : 8;
#pragma unroll
    for (int r = 0; r < 8; ++r) {
        int i = i0 + wm * 16 + mo + r;
        long base = ((long)b * N + i) * N + j0 + wn * 32 + c2;
        E[base]      = aR0[r] * wr + aD0[r] * wd;
        E[base + 16] = aR1[r] * wr + aD1[r] * wd;
    }
}

// ---------------------------------------------------------------------------
// Row softmax (N=1024) -> bf16 attention
// ---------------------------------------------------------------------------
__global__ void __launch_bounds__(256) softmax_kernel(
    const float* __restrict__ E, unsigned short* __restrict__ A)
{
    constexpr int N = 1024;
    const int row = blockIdx.x;
    const float* e = E + (long)row * N;
    unsigned short* a = A + (long)row * N;
    const int t = threadIdx.x;
    __shared__ float red[256];

    float mx = -3.4e38f;
    for (int j = t; j < N; j += 256) mx = fmaxf(mx, e[j]);
    red[t] = mx; __syncthreads();
    for (int s = 128; s > 0; s >>= 1) { if (t < s) red[t] = fmaxf(red[t], red[t + s]); __syncthreads(); }
    mx = red[0]; __syncthreads();

    float sum = 0.f;
    for (int j = t; j < N; j += 256) sum += __expf(e[j] - mx);
    red[t] = sum; __syncthreads();
    for (int s = 128; s > 0; s >>= 1) { if (t < s) red[t] += red[t + s]; __syncthreads(); }
    float inv = 1.f / red[0];

    for (int j = t; j < N; j += 256) a[j] = f2bf(__expf(e[j] - mx) * inv);
}

// ---------------------------------------------------------------------------
// out[b,c,i] = f2[b,c,i] + f3[b,c,i] + gamma * sum_j V[b,c,j]*Att[b,i,j]
// ---------------------------------------------------------------------------
__global__ void __launch_bounds__(256) av_wmma_kernel(
    const unsigned short* __restrict__ V,
    const unsigned short* __restrict__ At,
    const float* __restrict__ f2, const float* __restrict__ f3,
    const float* __restrict__ gamma,
    float* __restrict__ out, int C)
{
    constexpr int N = 1024;
    const int b  = blockIdx.z;
    const int c0 = blockIdx.y * 64, i0 = blockIdx.x * 64;
    const int t  = threadIdx.x;
    const int lane = t & 31, wv = t >> 5, wm = wv & 3, wn = wv >> 2;

    __shared__ alignas(16) unsigned short As[64][40];    // [c_local][k]
    __shared__ alignas(16) unsigned short Bs[16][136];   // pair-interleaved

    f32x8 acc0 = {0.f,0.f,0.f,0.f,0.f,0.f,0.f,0.f};
    f32x8 acc1 = acc0;

    const int ar = t >> 2, ac = (t & 3) << 3;
    const int bn2 = t >> 2, bkb = (t & 3) << 3;

    const int arow = wm * 16 + (lane & 15);
    const int aks  = (lane < 16) ? 0 : 16;
    const int bcol = (lane & 15) + wn * 32;
    const int bkb2 = (lane < 16) ? 0 : 8;

    for (int j0 = 0; j0 < N; j0 += 32) {
        __syncthreads();
        *(uint4*)&As[ar][ac] =
            *(const uint4*)(V + ((long)b * C + c0 + ar) * N + j0 + ac);
        {
            uint4 vv = *(const uint4*)(At + ((long)b * N + i0 + bn2) * N + j0 + bkb);
            const unsigned short* pv = (const unsigned short*)&vv;
#pragma unroll
            for (int j = 0; j < 8; ++j) {
                int k = bkb + j;
                Bs[k >> 1][(bn2 << 1) | (k & 1)] = pv[j];   // transpose + pair pack
            }
        }
        __syncthreads();

        u32x8 au, bu0, bu1;
#pragma unroll
        for (int v = 0; v < 8; ++v) {
            int k0 = (v << 1) + aks;
            au[v] = *(const unsigned*)&As[arow][k0];
        }
#pragma unroll
        for (int v = 0; v < 8; ++v) {
            bu0[v] = *(const unsigned*)&Bs[bkb2 + v][bcol << 1];
            bu1[v] = *(const unsigned*)&Bs[bkb2 + v][(bcol << 1) + 32];
        }
        bf16x16 a  = __builtin_bit_cast(bf16x16, au);
        bf16x16 b0 = __builtin_bit_cast(bf16x16, bu0);
        bf16x16 b1 = __builtin_bit_cast(bf16x16, bu1);
        acc0 = __builtin_amdgcn_wmma_f32_16x16x32_bf16(false, a, false, b0, (short)0, acc0, false, false);
        acc1 = __builtin_amdgcn_wmma_f32_16x16x32_bf16(false, a, false, b1, (short)0, acc1, false, false);
    }

    const float g = gamma[0];
    const int col = lane & 15, mo = (lane < 16) ? 0 : 8;
#pragma unroll
    for (int r = 0; r < 8; ++r) {
        int m = c0 + wm * 16 + mo + r;
        long o = ((long)b * C + m) * N + i0 + wn * 32 + col;
        out[o]      = f2[o]      + f3[o]      + g * acc0[r];
        out[o + 16] = f2[o + 16] + f3[o + 16] + g * acc1[r];
    }
}

// ---------------------------------------------------------------------------
// Small helper kernels  (H=W=32 -> HW=1024 shifts everywhere)
// ---------------------------------------------------------------------------
__global__ void cast_embed_kernel(const float* __restrict__ in,
                                  unsigned short* __restrict__ out,
                                  int C, int Ctot, int chOff, long total)
{
    long i = (long)blockIdx.x * 256 + threadIdx.x;
    if (i >= total) return;
    int hw = (int)(i & 1023);
    long bc = i >> 10;
    int c = (int)(bc % C);
    int b = (int)(bc / C);
    out[(((long)b * Ctot + chOff + c) << 10) + hw] = f2bf(in[i]);
}

__global__ void repack_kernel(const float* __restrict__ Wf,
                              unsigned short* __restrict__ Wb,
                              int Co, int Ci, int KHW, long total)
{
    long i = (long)blockIdx.x * 256 + threadIdx.x;
    if (i >= total) return;
    int tap = (int)(i % KHW);
    long r  = i / KHW;
    int ci  = (int)(r % Ci);
    int co  = (int)(r / Ci);
    Wb[((long)tap * Co + co) * Ci + ci] = f2bf(Wf[i]);
}

// mode 0: y = (conv+bias)*s + (b - m*s),  s = w/sqrt(v+eps)
// mode 1: y = conv + bias
__global__ void prep_kernel(const float* bnw, const float* bnb,
                            const float* bnm, const float* bnv,
                            const float* bias, float* sc, float* sh, int C, int mode)
{
    int c = blockIdx.x * 256 + threadIdx.x;
    if (c >= C) return;
    if (mode == 0) {
        float s = bnw[c] * rsqrtf(bnv[c] + 1e-5f);
        sc[c] = s;
        sh[c] = (bias ? bias[c] : 0.f) * s + bnb[c] - bnm[c] * s;
    } else {
        sc[c] = 1.f;
        sh[c] = bias ? bias[c] : 0.f;
    }
}

__global__ void mean_kernel(const unsigned short* __restrict__ X,
                            float* __restrict__ M)
{
    constexpr int HW = 1024;
    const int bc = blockIdx.x, t = threadIdx.x;
    __shared__ float red[256];
    float s = 0.f;
    for (int j = t; j < HW; j += 256) s += bf2f(X[((long)bc << 10) + j]);
    red[t] = s; __syncthreads();
    for (int k = 128; k > 0; k >>= 1) { if (t < k) red[t] += red[t + k]; __syncthreads(); }
    if (t == 0) M[bc] = red[0] * (1.f / HW);
}

__global__ void fc_kernel(const float* __restrict__ in, const float* __restrict__ Wt,
                          const float* __restrict__ sc, const float* __restrict__ sh,
                          float* __restrict__ out, int B, int Ci, int Co, int relu)
{
    int i = blockIdx.x * 256 + threadIdx.x;
    if (i >= B * Co) return;
    int b = i / Co, co = i % Co;
    float s = 0.f;
    for (int c = 0; c < Ci; ++c) s += in[b * Ci + c] * Wt[co * Ci + c];
    float v = s * sc[co] + sh[co];
    if (relu) v = fmaxf(v, 0.f);
    out[i] = v;
}

__global__ void contrast_kernel(const float* __restrict__ loc, const float* __restrict__ ctx,
                                const float* __restrict__ glb,
                                const float* __restrict__ s1, const float* __restrict__ t1,
                                const float* __restrict__ s2, const float* __restrict__ t2,
                                unsigned short* __restrict__ out,
                                int C, int Ctot, int chOff, long total)
{
    long i = (long)blockIdx.x * 256 + threadIdx.x;
    if (i >= total) return;
    int hw = (int)(i & 1023);
    long bc = i >> 10;
    int c = (int)(bc % C);
    int b = (int)(bc / C);
    float l = loc[i];
    float a = (l - ctx[i]) * s1[c] + t1[c];
    float d = (l - glb[b * C + c]) * s2[c] + t2[c];
    float v = fmaxf(a, 0.f) + fmaxf(d, 0.f);
    out[(((long)b * Ctot + chOff + c) << 10) + hw] = f2bf(v);
}

__global__ void softmax2_kernel(const float* lr, const float* ld, const float* gamma,
                                float* wsm, float* tail, int B)
{
    int b = threadIdx.x;
    if (b < B) {
        float a = lr[b], c = ld[b];
        float m = fmaxf(a, c);
        float ea = __expf(a - m), ec = __expf(c - m);
        float s = ea + ec;
        wsm[b]     = ea / s; wsm[B + b] = ec / s;
        tail[b]    = ea / s; tail[B + b] = ec / s;
    }
    if (b == 0) tail[2 * B] = gamma[0];
}

// ---------------------------------------------------------------------------
// Host orchestration
// ---------------------------------------------------------------------------
extern "C" void kernel_launch(void* const* d_in, const int* in_sizes, int n_in,
                              void* d_out, int out_size, void* d_ws, size_t ws_size,
                              hipStream_t stream)
{
    (void)in_sizes; (void)n_in; (void)out_size; (void)ws_size;
    const int B = 8, Cx = 256, Cy = 256, Cxy = 512, HW = 1024, N = 1024;
    const long NPIX = (long)B * HW;  // 8192

    const float* xin = (const float*)d_in[0];
    const float* yin = (const float*)d_in[1];
    auto F = [&](int i) { return (const float*)d_in[i]; };

    // ---- workspace carving (256B aligned) ----
    char* ws = (char*)d_ws;
    size_t off = 0;
    auto alloc = [&](size_t bytes) -> char* {
        off = (off + 255) & ~(size_t)255;
        char* p = ws + off; off += bytes; return p;
    };
    unsigned short* xb   = (unsigned short*)alloc((size_t)B * Cx * HW * 2);
    unsigned short* yb   = (unsigned short*)alloc((size_t)B * Cy * HW * 2);
    unsigned short* xyb  = (unsigned short*)alloc((size_t)B * Cxy * HW * 2);
    unsigned short* wpk  = (unsigned short*)alloc((size_t)7 * 1024 * 1024 * 2);
    unsigned short* f1b  = (unsigned short*)alloc((size_t)B * Cxy * HW * 2);
    float*          locf = (float*)alloc((size_t)B * Cxy * HW * 4);
    float*          ctxf = (float*)alloc((size_t)B * Cxy * HW * 4);
    unsigned short* catb = (unsigned short*)alloc((size_t)B * 1024 * HW * 2);
    float*          f2f  = (float*)alloc((size_t)B * Cxy * HW * 4);
    float*          f3f  = (float*)alloc((size_t)B * Cxy * HW * 4);
    unsigned short* f3b  = (unsigned short*)alloc((size_t)B * Cxy * HW * 2);
    unsigned short* vb   = (unsigned short*)alloc((size_t)B * Cxy * HW * 2);
    unsigned short* qrb  = (unsigned short*)alloc((size_t)B * 32 * HW * 2);
    unsigned short* krb  = (unsigned short*)alloc((size_t)B * 32 * HW * 2);
    unsigned short* qdb  = (unsigned short*)alloc((size_t)B * 32 * HW * 2);
    unsigned short* kdb  = (unsigned short*)alloc((size_t)B * 32 * HW * 2);
    float*          energy = (float*)alloc((size_t)B * N * N * 4);
    unsigned short* attb   = (unsigned short*)alloc((size_t)B * N * N * 2);
    float* st   = (float*)alloc(8192 * 4);
    float* sc0 = st,        *sh0 = st + 1024;
    float* sc1 = st + 2048, *sh1 = st + 3072;
    float* sc2 = st + 4096, *sh2 = st + 5120;
    float* mfus = (float*)alloc((size_t)B * 512 * 4);
    float* mx   = (float*)alloc((size_t)B * 256 * 4);
    float* my   = (float*)alloc((size_t)B * 256 * 4);
    float* glb  = (float*)alloc((size_t)B * 512 * 4);
    float* hid  = (float*)alloc((size_t)B * 32 * 4);
    float* logr = (float*)alloc((size_t)B * 4);
    float* logd = (float*)alloc((size_t)B * 4);
    float* wsm  = (float*)alloc(16 * 4);

    // ---- launch helpers ----
    auto cast_embed = [&](const float* src, unsigned short* dst, int C, int Ctot, int chOff) {
        long total = (long)B * C * HW;
        cast_embed_kernel<<<dim3((unsigned)((total + 255) / 256)), dim3(256), 0, stream>>>(
            src, dst, C, Ctot, chOff, total);
    };
    auto prep = [&](const float* bw, const float* bb, const float* bm, const float* bv,
                    const float* bias, float* s, float* t, int C, int mode) {
        prep_kernel<<<dim3((C + 255) / 256), dim3(256), 0, stream>>>(bw, bb, bm, bv, bias, s, t, C, mode);
    };
    auto repack = [&](const float* Wf, int Co, int Ci, int KHW) {
        long total = (long)Co * Ci * KHW;
        repack_kernel<<<dim3((unsigned)((total + 255) / 256)), dim3(256), 0, stream>>>(
            Wf, wpk, Co, Ci, KHW, total);
    };
    auto conv = [&](const unsigned short* X, const float* s, const float* t,
                    float* Yf, unsigned short* Yb, int Cin, int Cout, int ks, int relu) {
        dim3 g((unsigned)(NPIX / 64), (unsigned)((Cout + 63) / 64));
        if (ks == 3)
            conv_wmma_kernel<3, 1, 1><<<g, dim3(256), 0, stream>>>(X, wpk, s, t, Yf, Yb, Cin, Cout, relu);
        else if (ks == 5)
            conv_wmma_kernel<5, 4, 2><<<g, dim3(256), 0, stream>>>(X, wpk, s, t, Yf, Yb, Cin, Cout, relu);
        else
            conv_wmma_kernel<1, 0, 1><<<g, dim3(256), 0, stream>>>(X, wpk, s, t, Yf, Yb, Cin, Cout, relu);
    };
    auto cbr = [&](int p, const unsigned short* X, int Cin, int Cout,
                   int ks, float* Yf, unsigned short* Yb) {
        prep(F(p + 2), F(p + 3), F(p + 4), F(p + 5), F(p + 1), sc0, sh0, Cout, 0);
        repack(F(p), Cout, Cin, ks * ks);
        conv(X, sc0, sh0, Yf, Yb, Cin, Cout, ks, 1);
    };
    auto mean = [&](const unsigned short* X, float* M, int C) {
        mean_kernel<<<dim3(B * C), dim3(256), 0, stream>>>(X, M);
    };
    auto fc = [&](const float* in, const float* Wt, const float* s, const float* t,
                  float* outp, int Ci, int Co, int relu) {
        int n = B * Co;
        fc_kernel<<<dim3((n + 255) / 256), dim3(256), 0, stream>>>(in, Wt, s, t, outp, B, Ci, Co, relu);
    };
    auto contrast = [&](const float* lo, const float* cx, const float* gl, int C, int chOff) {
        long total = (long)B * C * HW;
        contrast_kernel<<<dim3((unsigned)((total + 255) / 256)), dim3(256), 0, stream>>>(
            lo, cx, gl, sc1, sh1, sc2, sh2, catb, C, 1024, chOff, total);
    };

    // ---- 1) inputs -> bf16 (x, y, and concatenated xy) ----
    cast_embed(xin, xb, Cx, Cx, 0);
    cast_embed(yin, yb, Cy, Cy, 0);
    cast_embed(xin, xyb, Cx, Cxy, 0);
    cast_embed(yin, xyb, Cy, Cxy, Cx);

    // ---- 2) fusion1 (bf16 out only) ----
    cbr(2, xyb, Cxy, Cxy, 3, nullptr, f1b);

    // ---- 3) main branch contrast ----
    cbr(8,  f1b, Cxy, Cxy, 3, locf, nullptr);                // local_main
    cbr(14, f1b, Cxy, Cxy, 5, ctxf, nullptr);                // context_main (dil=2)
    mean(f1b, mfus, Cxy);
    prep(F(22), F(23), F(24), F(25), F(21), sc0, sh0, Cxy, 0);
    fc(mfus, F(20), sc0, sh0, glb, Cxy, Cxy, 1);             // global_main
    prep(F(26), F(27), F(28), F(29), nullptr, sc1, sh1, Cxy, 0);  // bn_main1
    prep(F(30), F(31), F(32), F(33), nullptr, sc2, sh2, Cxy, 0);  // bn_main2
    contrast(locf, ctxf, glb, Cxy, 0);

    // ---- 4) rgb branch contrast ----
    cbr(34, xb, Cx, Cx, 3, locf, nullptr);
    cbr(40, xb, Cx, Cx, 5, ctxf, nullptr);
    mean(xb, mx, Cx);
    prep(F(48), F(49), F(50), F(51), F(47), sc0, sh0, Cx, 0);
    fc(mx, F(46), sc0, sh0, glb, Cx, Cx, 1);
    prep(F(52), F(53), F(54), F(55), nullptr, sc1, sh1, Cx, 0);
    prep(F(56), F(57), F(58), F(59), nullptr, sc2, sh2, Cx, 0);
    contrast(locf, ctxf, glb, Cx, 512);

    // ---- 5) depth branch contrast ----
    cbr(60, yb, Cy, Cy, 3, locf, nullptr);
    cbr(66, yb, Cy, Cy, 5, ctxf, nullptr);
    mean(yb, my, Cy);
    prep(F(74), F(75), F(76), F(77), F(73), sc0, sh0, Cy, 0);
    fc(my, F(72), sc0, sh0, glb, Cy, Cy, 1);
    prep(F(78), F(79), F(80), F(81), nullptr, sc1, sh1, Cy, 0);
    prep(F(82), F(83), F(84), F(85), nullptr, sc2, sh2, Cy, 0);
    contrast(locf, ctxf, glb, Cy, 768);

    // ---- 6) fusion2, fusion3 ----
    cbr(86, catb, 1024, Cxy, 3, f2f, nullptr);
    cbr(92, xyb,  Cxy,  Cxy, 3, f3f, f3b);

    // ---- 7) value = 1x1 conv (bias only) ----
    prep(nullptr, nullptr, nullptr, nullptr, F(99), sc0, sh0, Cxy, 1);
    repack(F(98), Cxy, Cxy, 1);
    conv(f3b, sc0, sh0, nullptr, vb, Cxy, Cxy, 1, 0);

    // ---- 8) MLP gates + softmax over {rgb, depth} ----
    prep(F(102), F(103), F(104), F(105), F(101), sc0, sh0, 32, 0);
    fc(mx, F(100), sc0, sh0, hid, Cx, 32, 1);
    prep(nullptr, nullptr, nullptr, nullptr, F(107), sc0, sh0, 1, 1);
    fc(hid, F(106), sc0, sh0, logr, 32, 1, 0);
    prep(F(110), F(111), F(112), F(113), F(109), sc0, sh0, 32, 0);
    fc(my, F(108), sc0, sh0, hid, Cy, 32, 1);
    prep(nullptr, nullptr, nullptr, nullptr, F(115), sc0, sh0, 1, 1);
    fc(hid, F(114), sc0, sh0, logd, 32, 1, 0);

    float* tail = (float*)d_out + (long)B * Cxy * HW;   // w_rgb[8], w_depth[8], gamma[1]
    softmax2_kernel<<<dim3(1), dim3(32), 0, stream>>>(logr, logd, F(124), wsm, tail, B);

    // ---- 9) q/k 1x1 projections (Cout=32) ----
    auto proj = [&](int p, const unsigned short* X, unsigned short* outp) {
        prep(nullptr, nullptr, nullptr, nullptr, F(p + 1), sc0, sh0, 32, 1);
        repack(F(p), 32, Cx, 1);
        conv(X, sc0, sh0, nullptr, outp, Cx, 32, 1, 0);
    };
    proj(116, xb, qrb);
    proj(118, xb, krb);
    proj(120, yb, qdb);
    proj(122, yb, kdb);

    // ---- 10) energy -> softmax -> attention apply (+ final fused add) ----
    qk_wmma_kernel<<<dim3(16, 16, 8), dim3(256), 0, stream>>>(qrb, krb, qdb, kdb, wsm, energy);
    softmax_kernel<<<dim3(B * N), dim3(256), 0, stream>>>(energy, attb);
    av_wmma_kernel<<<dim3(16, 8, 8), dim3(256), 0, stream>>>(
        vb, attb, f2f, f3f, F(124), (float*)d_out, Cxy);
}
